// ResidualGraph_22960895164951
// MI455X (gfx1250) — matrix operand order repaired and checked by reading, not compiled, not run on hardware
//
#include <hip/hip_runtime.h>
#include <hip/hip_bf16.h>
#include <cstdint>

typedef __attribute__((ext_vector_type(16))) __bf16 v16bf;
typedef __attribute__((ext_vector_type(8)))  float  v8f;
typedef __attribute__((ext_vector_type(4)))  unsigned int v4u;
typedef __attribute__((ext_vector_type(8)))  int    v8i;
typedef __attribute__((ext_vector_type(4)))  int    v4i;

namespace {
constexpr int CV       = 62;   // valid graph nodes
constexpr int CP       = 64;   // padded node count
constexpr int FIN      = 256;  // input features
constexpr int NF       = 64;   // hidden / output features
constexpr int KTOP     = 10;
constexpr int NTHREADS = 256;  // 8 wave32 waves
constexpr int NWAVES   = NTHREADS / 32;
}

// ---------------------------------------------------------------------------
// WMMA tile helpers (CDNA5 wave32 16x16x32 bf16 layouts, ISA 7.12.2)
// ---------------------------------------------------------------------------
__device__ __forceinline__ v16bf load_A_bf(const __bf16* A, int lda, int row0,
                                           int k0, int lane) {
  const int m  = row0 + (lane & 15);
  const int kb = (lane & 16) ? 8 : 0;
  const __bf16* p0 = A + m * lda + k0 + kb;       // K = k0+kb .. +7
  const __bf16* p1 = p0 + 16;                     // K = k0+16+kb .. +7
  v16bf a;
#pragma unroll
  for (int j = 0; j < 8; ++j) { a[j] = p0[j]; a[j + 8] = p1[j]; }
  return a;
}

__device__ __forceinline__ v16bf load_B_bf(const __bf16* Bt, int ldb, int n0,
                                           int k0, int lane) {
  const int n  = n0 + (lane & 15);
  const int kb = (lane & 16) ? 8 : 0;
  const __bf16* p = Bt + n * ldb + k0 + kb;
  v16bf b;
#pragma unroll
  for (int j = 0; j < 8; ++j) { b[j] = p[j]; b[j + 8] = p[j + 16]; }
  return b;
}

__device__ __forceinline__ v8f tile_mm(const __bf16* A, int lda,
                                       const __bf16* Bt, int ldb,
                                       int row0, int n0, int ksteps, int lane,
                                       v8f acc) {
  for (int kk = 0; kk < ksteps; ++kk) {
    v16bf a = load_A_bf(A, lda, row0, kk * 32, lane);
    v16bf b = load_B_bf(Bt, ldb, n0, kk * 32, lane);
    acc = __builtin_amdgcn_wmma_f32_16x16x32_bf16(false, a, false, b,
                                                  (short)0, acc, false, false);
  }
  return acc;
}

// Stage a [K x N] f32 global weight into LDS transposed bf16 Bt[n][k].
__device__ __forceinline__ void stage_WT(const float* __restrict__ W,
                                         __bf16* Wt, int Kdim, int Ndim,
                                         int tid) {
  const int total = Kdim * Ndim;
  for (int idx = tid; idx < total; idx += NTHREADS) {
    const int k = idx / Ndim;
    const int n = idx - k * Ndim;
    Wt[n * Kdim + k] = (__bf16)W[idx];
  }
}

// Branchless fast tanh: (e^2x - 1) / (e^2x + 1), clamped against overflow.
__device__ __forceinline__ float fast_tanh(float x) {
  const float xc = fminf(fmaxf(x, -15.f), 15.f);
  const float e  = __expf(2.f * xc);
  return __fdividef(e - 1.f, e + 1.f);
}

// ---------------------------------------------------------------------------
// One batch element per workgroup. 8 waves run the whole pipeline out of LDS.
// ---------------------------------------------------------------------------
__global__ void __launch_bounds__(NTHREADS, 1)
residual_graph_kernel(const float* __restrict__ x,
                      const float* __restrict__ gate_w,
                      const float* __restrict__ gate_b,
                      const float* __restrict__ bnlin_w,
                      const float* __restrict__ bnlin_b,
                      const float* __restrict__ rel_w1,
                      const float* __restrict__ rel_b1,
                      const float* __restrict__ root_w1,
                      const float* __restrict__ rel_w_mid,
                      const float* __restrict__ rel_b_mid,
                      const float* __restrict__ root_w_mid,
                      float* __restrict__ out) {
  __shared__ __bf16 s_xs[CP * FIN];    // x (bf16, row-major, pad rows zero)
  __shared__ __bf16 s_wt[NF * FIN];    // staged transposed weight(s)
  __shared__ __bf16 s_xa[CP * NF];     // xa (A-role AND Bt-role for xa@xa^T)
  __shared__ __bf16 s_adjb[CP * CP];   // masked adjacency, bf16 (A-role)
  __shared__ __bf16 s_pt[NF * CP];     // p = h@W_rel, stored transposed (Bt)
  __shared__ __bf16 s_hb[CP * NF];     // h, bf16 row-major (A-role)
  // f32 work buffers (adjf/xm/q/h) are dead during stage 0, so the TDM f32
  // staging area for x overlays them: 64 KB union, total LDS stays 160 KB.
  __shared__ union {
    float stage_x[CP * FIN];           // raw f32 x tile landed by the TDM
    struct {
      float adjf[CP * CP];             // scores / softmax adjacency
      float xm[CP * NF];               // gate mask (tanh)
      float q[CP * NF];                // q = h@W_root
      float h[CP * NF];                // h (residual accumulator)
    } w;
  } uf;

  const int b    = blockIdx.x;
  const int tid  = threadIdx.x;
  const int lane = tid & 31;
  const int wave = tid >> 5;

  const float* xg = x + (size_t)b * CV * FIN;

  // ---- Stage 0a: TDM DMA of x (wave 0) || stage gate_w^T (waves 1-7) -----
  if (wave == 0) {
    // Tensor DMA descriptor (ISA 8.3/8.4): 2D tensor, 62x256 f32 full tile.
    const uint64_t ga = (uint64_t)(uintptr_t)xg;
    const uint32_t la = (uint32_t)(uintptr_t)(&uf.stage_x[0]); // LDS byte addr
    v4u g0;
    g0[0] = 1u;                                   // count=1, no gather
    g0[1] = la;                                   // lds_addr[31:0]
    g0[2] = (uint32_t)(ga & 0xFFFFFFFFu);         // global_addr[31:0]
    g0[3] = (uint32_t)((ga >> 32) & 0x01FFFFFFu)  // global_addr[56:32]
          | (2u << 30);                           // type = 2 ("image")
    v8i g1;
    g1[0] = (int)(2u << 16);                      // data_size=2 -> 4 bytes
    g1[1] = (int)((uint32_t)FIN << 16);           // tensor_dim0[15:0]
    g1[2] = (int)((uint32_t)CV << 16);            // tensor_dim0[31:16]=0 | tensor_dim1[15:0]
    g1[3] = (int)((uint32_t)FIN << 16);           // tensor_dim1[31:16]=0 | tile_dim0=256
    g1[4] = CV;                                   // tile_dim1=62, tile_dim2=0
    g1[5] = FIN;                                  // tensor_dim0_stride[31:0]
    g1[6] = 0;                                    // stride hi / dim1_stride lo
    g1[7] = 0;                                    // dim1_stride hi
    v4i gz4 = {0, 0, 0, 0};                       // groups 2/3 unused (2D)
    v8i gz8 = {0, 0, 0, 0, 0, 0, 0, 0};           // extra group (clang-23 ABI)
    __builtin_amdgcn_tensor_load_to_lds(g0, g1, gz4, gz4, gz8, 0);
    __builtin_amdgcn_s_wait_tensorcnt(0);
  } else {
    const int t2 = tid - 32;                      // 224 staging threads
    for (int idx = t2; idx < FIN * NF; idx += (NTHREADS - 32)) {
      const int k = idx >> 6, n = idx & 63;
      s_wt[n * FIN + k] = (__bf16)gate_w[idx];
    }
  }
  __syncthreads();

  // ---- Stage 0b: convert staged f32 x -> bf16 (pad rows 62,63 zero) ------
  for (int idx = tid; idx < CP * FIN; idx += NTHREADS) {
    const int m = idx >> 8;                       // FIN = 256
    const float v = (m < CV) ? uf.stage_x[idx] : 0.f;
    s_xs[idx] = (__bf16)v;
  }
  __syncthreads();

  // ---- Stage 1: x_mask = tanh(x @ gate_w + gate_b) -> xm -----------------
  for (int t = wave; t < 16; t += NWAVES) {
    const int r0 = (t >> 2) << 4, n0 = (t & 3) << 4;
    v8f acc = {};
    acc = tile_mm(s_xs, FIN, s_wt, FIN, r0, n0, FIN / 32, lane, acc);
    const int n  = n0 + (lane & 15);
    const int mb = r0 + ((lane & 16) ? 8 : 0);
    const float bias = gate_b[n];
#pragma unroll
    for (int r = 0; r < 8; ++r)
      uf.w.xm[(mb + r) * NF + n] = fast_tanh(acc[r] + bias);
  }
  __syncthreads();

  // ---- Stage 2: xa = tanh(x @ bnlin_w + bnlin_b) (pad rows 0) ------------
  stage_WT(bnlin_w, s_wt, FIN, NF, tid);
  __syncthreads();
  for (int t = wave; t < 16; t += NWAVES) {
    const int r0 = (t >> 2) << 4, n0 = (t & 3) << 4;
    v8f acc = {};
    acc = tile_mm(s_xs, FIN, s_wt, FIN, r0, n0, FIN / 32, lane, acc);
    const int n  = n0 + (lane & 15);
    const int mb = r0 + ((lane & 16) ? 8 : 0);
    const float bias = bnlin_b[n];
#pragma unroll
    for (int r = 0; r < 8; ++r) {
      const int m = mb + r;
      const float v = (m < CV) ? fast_tanh(acc[r] + bias) : 0.f;
      s_xa[m * NF + n] = (__bf16)v;
    }
  }
  __syncthreads();

  // ---- Stage 3: scores = xa @ xa^T  (Bt of xa^T is xa itself) ------------
  for (int t = wave; t < 16; t += NWAVES) {
    const int r0 = (t >> 2) << 4, n0 = (t & 3) << 4;
    v8f acc = {};
    acc = tile_mm(s_xa, NF, s_xa, NF, r0, n0, NF / 32, lane, acc);
    const int n  = n0 + (lane & 15);
    const int mb = r0 + ((lane & 16) ? 8 : 0);
#pragma unroll
    for (int r = 0; r < 8; ++r) uf.w.adjf[(mb + r) * CP + n] = acc[r];
  }
  __syncthreads();

  // ---- Stage 4: row softmax + top-10 threshold mask -> s_adjb ------------
  if (tid < CP) {
    __bf16* arow = s_adjb + tid * CP;
    if (tid < CV) {
      float* row = uf.w.adjf + tid * CP;
      float mx = -1e30f;
      for (int n = 0; n < CV; ++n) mx = fmaxf(mx, row[n]);
      float ssum = 0.f;
      for (int n = 0; n < CV; ++n) { const float e = __expf(row[n] - mx); row[n] = e; ssum += e; }
      const float inv = 1.f / ssum;
      for (int n = 0; n < CV; ++n) row[n] *= inv;
      float thr = 1e30f;                          // KTOP-th largest
      for (int it = 0; it < KTOP; ++it) {
        float cur = -1e30f;
        for (int n = 0; n < CV; ++n) {
          const float v = row[n];
          if (v < thr && v > cur) cur = v;
        }
        thr = cur;
      }
      for (int n = 0; n < CP; ++n) {
        const float v = (n < CV && row[n] >= thr) ? row[n] : 0.f;
        arow[n] = (__bf16)v;
      }
    } else {
      for (int n = 0; n < CP; ++n) arow[n] = (__bf16)0.f;
    }
  }
  __syncthreads();

  // ---- Stage 5: layer 1.  adj@(x@Wr) + b + x@Wo --------------------------
  stage_WT(rel_w1, s_wt, FIN, NF, tid);
  __syncthreads();
  for (int t = wave; t < 16; t += NWAVES) {               // p = x@rel_w1 -> pt
    const int r0 = (t >> 2) << 4, n0 = (t & 3) << 4;
    v8f acc = {};
    acc = tile_mm(s_xs, FIN, s_wt, FIN, r0, n0, FIN / 32, lane, acc);
    const int n  = n0 + (lane & 15);
    const int mb = r0 + ((lane & 16) ? 8 : 0);
#pragma unroll
    for (int r = 0; r < 8; ++r) {
      const int m = mb + r;
      s_pt[n * CP + m] = (__bf16)((m < CV) ? acc[r] : 0.f); // transposed (Bt)
    }
  }
  __syncthreads();
  stage_WT(root_w1, s_wt, FIN, NF, tid);
  __syncthreads();
  for (int t = wave; t < 16; t += NWAVES) {               // q = x@root_w1
    const int r0 = (t >> 2) << 4, n0 = (t & 3) << 4;
    v8f acc = {};
    acc = tile_mm(s_xs, FIN, s_wt, FIN, r0, n0, FIN / 32, lane, acc);
    const int n  = n0 + (lane & 15);
    const int mb = r0 + ((lane & 16) ? 8 : 0);
#pragma unroll
    for (int r = 0; r < 8; ++r) {
      const int m = mb + r;
      uf.w.q[m * NF + n] = (m < CV) ? acc[r] : 0.f;
    }
  }
  __syncthreads();
  for (int t = wave; t < 16; t += NWAVES) {               // h1 = relu(adj@p+q+b)
    const int r0 = (t >> 2) << 4, n0 = (t & 3) << 4;
    v8f acc = {};
    acc = tile_mm(s_adjb, CP, s_pt, CP, r0, n0, CP / 32, lane, acc);
    const int n  = n0 + (lane & 15);
    const int mb = r0 + ((lane & 16) ? 8 : 0);
    const float bias = rel_b1[n];
#pragma unroll
    for (int r = 0; r < 8; ++r) {
      const int m = mb + r;
      const float v =
          (m < CV) ? fmaxf(acc[r] + uf.w.q[m * NF + n] + bias, 0.f) : 0.f;
      uf.w.h[m * NF + n] = v;
      s_hb[m * NF + n]   = (__bf16)v;
    }
  }
  __syncthreads();

  // ---- Stage 6: layers 2..7 (L<5 residual, L==5 plain) -------------------
  for (int L = 0; L < 6; ++L) {
    const float* wr = rel_w_mid + (size_t)L * NF * NF;
    const float* wo = root_w_mid + (size_t)L * NF * NF;
    const float* bm = rel_b_mid + (size_t)L * NF;
    for (int idx = tid; idx < NF * NF; idx += NTHREADS) { // stage wr^T, wo^T
      const int k = idx >> 6, n = idx & 63;
      s_wt[n * NF + k]           = (__bf16)wr[idx];
      s_wt[NF * NF + n * NF + k] = (__bf16)wo[idx];
    }
    __syncthreads();
    for (int t = wave; t < 32; t += NWAVES) {  // p = h@wr -> pt ; q = h@wo
      const int which = t >> 4;                // wave-uniform per iteration
      const int tt = t & 15;
      const int r0 = (tt >> 2) << 4, n0 = (tt & 3) << 4;
      v8f acc = {};
      acc = tile_mm(s_hb, NF, s_wt + which * NF * NF, NF, r0, n0, NF / 32,
                    lane, acc);
      const int n  = n0 + (lane & 15);
      const int mb = r0 + ((lane & 16) ? 8 : 0);
      if (which == 0) {
#pragma unroll
        for (int r = 0; r < 8; ++r) s_pt[n * CP + (mb + r)] = (__bf16)acc[r];
      } else {
#pragma unroll
        for (int r = 0; r < 8; ++r) uf.w.q[(mb + r) * NF + n] = acc[r];
      }
    }
    __syncthreads();
    for (int t = wave; t < 16; t += NWAVES) {  // h' = relu(adj@p + q + b) [+h]
      const int r0 = (t >> 2) << 4, n0 = (t & 3) << 4;
      v8f acc = {};
      acc = tile_mm(s_adjb, CP, s_pt, CP, r0, n0, CP / 32, lane, acc);
      const int n  = n0 + (lane & 15);
      const int mb = r0 + ((lane & 16) ? 8 : 0);
      const float bias = bm[n];
#pragma unroll
      for (int r = 0; r < 8; ++r) {
        const int m = mb + r;
        float v = fmaxf(acc[r] + uf.w.q[m * NF + n] + bias, 0.f);
        if (L < 5) v += uf.w.h[m * NF + n];
        v = (m < CV) ? v : 0.f;
        uf.w.h[m * NF + n] = v;
        s_hb[m * NF + n]   = (__bf16)v;
      }
    }
    __syncthreads();
  }

  // ---- Stage 7: out = h * x_mask -----------------------------------------
  float* og = out + (size_t)b * CV * NF;
  for (int idx = tid; idx < CV * NF; idx += NTHREADS)
    og[idx] = uf.w.h[idx] * uf.w.xm[idx];  // flat layouts coincide for m < CV
}

// ---------------------------------------------------------------------------
extern "C" void kernel_launch(void* const* d_in, const int* in_sizes, int n_in,
                              void* d_out, int out_size, void* d_ws,
                              size_t ws_size, hipStream_t stream) {
  const float* x        = (const float*)d_in[0];
  const float* gate_w   = (const float*)d_in[1];
  const float* gate_b   = (const float*)d_in[2];
  const float* bnlin_w  = (const float*)d_in[3];
  const float* bnlin_b  = (const float*)d_in[4];
  const float* rel_w1   = (const float*)d_in[5];
  const float* rel_b1   = (const float*)d_in[6];
  const float* root_w1  = (const float*)d_in[7];
  const float* rel_wm   = (const float*)d_in[8];
  const float* rel_bm   = (const float*)d_in[9];
  const float* root_wm  = (const float*)d_in[10];
  float* out            = (float*)d_out;

  const int Bn = in_sizes[0] / (CV * FIN);   // 4096
  residual_graph_kernel<<<dim3(Bn), dim3(NTHREADS), 0, stream>>>(
      x, gate_w, gate_b, bnlin_w, bnlin_b, rel_w1, rel_b1, root_w1,
      rel_wm, rel_bm, root_wm, out);
}
